// LSTMDQN_88957362635379
// MI455X (gfx1250) — matrix operand order repaired and checked by reading, not compiled, not run on hardware
//
#include <hip/hip_runtime.h>
#include <hip/hip_bf16.h>
#include <math.h>

// ---------------------------------------------------------------------------
// LSTM-DQN fused kernel for gfx1250 (MI455X), wave32 WMMA bf16.
// B=16384, S=30, E=H=100, 4H=400 gates, heads [100,6]x2.
//
// K index space (padded): 0..111 = x (embed, real 0..99), 112..223 = h
// (real 0..99).  7 K-chunks of 32 for v_wmma_f32_16x16x32_bf16.
// N per gate padded 100 -> 112 (7 u-groups of 16).
// Inner matmul: fully-unrolled g(7) x gate(4) x kc(7) nest; B tiles use a
// 3-slot buffer indexed by compile-time (n % 3) with tile n+2 prefetched
// before the WMMA of tile n -> partial s_wait_dscnt, latency hidden.
// ---------------------------------------------------------------------------

typedef __bf16 bf16_t;
typedef __attribute__((ext_vector_type(16))) __bf16 v16bf;
typedef __attribute__((ext_vector_type(8)))  float  v8f;

#define B_TOT 16384
#define SEQ   30
#define EMB   100
#define HID   100
#define KPAD  224
#define UPAD  112
#define NGATE 4
#define WAVES 8
#define ROWS_PER_BLOCK (WAVES * 16)
#define NTILE 196   // 7 groups * 4 gates * 7 k-chunks

union V16 { v16bf v; uint4 q[2]; };

// Fast gate nonlinearities: v_exp_f32 + v_rcp_f32 (correct saturation at +-inf).
__device__ __forceinline__ float sigmoid_fast(float x) {
    return __builtin_amdgcn_rcpf(1.0f + __expf(-x));
}
__device__ __forceinline__ float tanh_fast(float x) {
    return 1.0f - 2.0f * __builtin_amdgcn_rcpf(__expf(2.0f * x) + 1.0f);
}

__global__ __launch_bounds__(256, 1)
void lstm_dqn_kernel(const int*   __restrict__ inputs,   // [B, S] int32
                     const float* __restrict__ embed,    // [100, 100]
                     const float* __restrict__ W_lstm,   // [200, 400]
                     const float* __restrict__ b_lstm,   // [400]
                     const float* __restrict__ W_act,    // [100, 6]
                     const float* __restrict__ b_act,    // [6]
                     const float* __restrict__ W_obj,    // [100, 6]
                     const float* __restrict__ b_obj,    // [6]
                     float*       __restrict__ out)      // [B*6 reward | B*6 object]
{
    // ---- LDS (~248 KB of the 320 KB WGP budget) ----
    __shared__ bf16_t sWt[NGATE][UPAD][KPAD];   // W^T, bf16, zero-padded: 200704 B
    __shared__ bf16_t sEmb[EMB * UPAD];         // embedding rows, padded: 22400 B
    __shared__ float  sBias[NGATE * UPAD];      // 1792 B
    __shared__ bf16_t sH[WAVES][16][UPAD];      // per-wave h state: 28672 B

    const int tid = threadIdx.x;

    // ---------------- one-time LDS fills ----------------
    for (int idx = tid; idx < NGATE * UPAD * KPAD; idx += 256) {
        int g = idx / (UPAD * KPAD);
        int r = idx % (UPAD * KPAD);
        int u = r / KPAD;
        int k = r % KPAD;
        float v = 0.0f;
        if (u < HID) {
            if (k < 112) { if (k < EMB) v = W_lstm[k * 400 + g * HID + u]; }
            else { int kk = k - 112; if (kk < HID) v = W_lstm[(EMB + kk) * 400 + g * HID + u]; }
        }
        (&sWt[0][0][0])[idx] = (bf16_t)v;
    }
    for (int idx = tid; idx < EMB * UPAD; idx += 256) {
        int r = idx / UPAD, c = idx % UPAD;
        sEmb[idx] = (bf16_t)((c < EMB) ? embed[r * EMB + c] : 0.0f);
    }
    for (int idx = tid; idx < NGATE * UPAD; idx += 256) {
        int g = idx / UPAD, u = idx % UPAD;
        sBias[idx] = (u < HID) ? b_lstm[g * HID + u] : 0.0f;
    }
    {   // zero h state
        int* hz = (int*)&sH[0][0][0];
        for (int idx = tid; idx < (int)(sizeof(sH) / 4); idx += 256) hz[idx] = 0;
    }
    __syncthreads();

    // ---------------- per-wave setup ----------------
    const int wv   = tid >> 5;
    const int lane = tid & 31;
    const int half = lane >> 4;        // 0: lanes 0-15, 1: lanes 16-31
    const int l16  = lane & 15;
    const int hsel = half * 8;         // A-matrix K sub-select per lane half
    const int b0   = blockIdx.x * ROWS_PER_BLOCK + wv * 16;
    const int row  = b0 + l16;         // this lane's batch row (A-matrix M = l16)

    // pack this row's 30 tokens (each < 100) into 8 dwords
    unsigned tokpack[8];
#pragma unroll
    for (int w = 0; w < 8; ++w) {
        unsigned p = 0;
#pragma unroll
        for (int j = 0; j < 4; ++j) {
            int t = w * 4 + j;
            if (t < SEQ) p |= ((unsigned)inputs[row * SEQ + t] & 0xffu) << (8 * j);
        }
        tokpack[w] = p;
    }

    float c_state[7][8];
    float hsum[7][8];
#pragma unroll
    for (int g = 0; g < 7; ++g)
#pragma unroll
        for (int r = 0; r < 8; ++r) { c_state[g][r] = 0.0f; hsum[g][r] = 0.0f; }

    const bf16_t* embp  = &sEmb[0];
    const bf16_t* hp    = &sH[wv][l16][0];              // A-matrix reads (row = l16)
    bf16_t*       hwl   = &sH[wv][0][0] + (half ? 8 : 0) * UPAD + l16;  // D writes
    const float*  biasl = &sBias[l16];
    const bf16_t* wlane = &sWt[0][0][0] + l16 * KPAD + half * 8;        // B base

    // Linear tile n = g*28 + gate*7 + kc (all compile-time in the unrolled nest).
#define TILE_G(n)    ((n) / 28)
#define TILE_GATE(n) (((n) % 28) / 7)
#define TILE_KC(n)   ((n) % 7)
#define LDB_N(n, dst) do {                                                    \
        const bf16_t* p_ = wlane +                                            \
            (TILE_GATE(n) * UPAD + TILE_G(n) * 16) * KPAD + TILE_KC(n) * 32;  \
        (dst).q[0] = *(const uint4*)p_;                                       \
        (dst).q[1] = *(const uint4*)(p_ + 16);                                \
    } while (0)

    // ---------------- recurrent loop ----------------
#pragma unroll 1
    for (int t = 0; t < SEQ; ++t) {
        // Stop LICM from hoisting the (loop-invariant) LDS W-tile loads out of
        // this 30-iteration loop and spilling them to scratch.
        asm volatile("" ::: "memory");

        const int tok = (tokpack[t >> 2] >> ((t & 3) * 8)) & 0xff;

        // ---- load all 7 A K-chunks (concat(x_t, h_{t-1}), 16x32 each) ----
        V16 a[7];
#pragma unroll
        for (int c = 0; c < 3; ++c) {                 // x: K 0..95
            int o0 = tok * UPAD + c * 32 + hsel;
            a[c].q[0] = *(const uint4*)(embp + o0);
            a[c].q[1] = *(const uint4*)(embp + o0 + 16);
        }
        {   // chunk 3: K 96..111 from x, K 112..127 from h
            a[3].q[0] = *(const uint4*)(embp + tok * UPAD + 96 + hsel);
            a[3].q[1] = *(const uint4*)(hp + hsel);
        }
#pragma unroll
        for (int c = 4; c < 7; ++c) {                 // h: K 128..223
            int hc0 = (c * 32 - 112) + hsel;
            a[c].q[0] = *(const uint4*)(hp + hc0);
            a[c].q[1] = *(const uint4*)(hp + hc0 + 16);
        }

        // ---- 196-tile WMMA stream, 3-slot B buffers, prefetch distance 2 ----
        V16 bb[3];
        LDB_N(0, bb[0]);
        LDB_N(1, bb[1]);
        v8f accs[4];
#pragma unroll
        for (int g = 0; g < 7; ++g) {
#pragma unroll
            for (int gate = 0; gate < NGATE; ++gate) {
#pragma unroll
                for (int kc = 0; kc < 7; ++kc) {
                    const int n = g * 28 + gate * 7 + kc;   // compile-time
                    if (n + 2 < NTILE) LDB_N(n + 2, bb[(n + 2) % 3]);
                    const v8f zero = {0.f, 0.f, 0.f, 0.f, 0.f, 0.f, 0.f, 0.f};
                    accs[gate] = __builtin_amdgcn_wmma_f32_16x16x32_bf16(
                        false, a[kc].v, false, bb[n % 3].v, (short)0,
                        (kc == 0) ? zero : accs[gate], false, false);
                }
            }

            // ---- gate tail for group g (co-executes with XDL WMMA) ----
            const float bi = biasl[0 * UPAD + g * 16];
            const float bj = biasl[1 * UPAD + g * 16];
            const float bf = biasl[2 * UPAD + g * 16];
            const float bo = biasl[3 * UPAD + g * 16];
#pragma unroll
            for (int r = 0; r < 8; ++r) {
                // D layout: VGPR r -> row (half? r+8 : r), lane -> col g*16+l16
                float ig = sigmoid_fast(accs[0][r] + bi);
                float jg = tanh_fast(accs[1][r] + bj);
                float fg = sigmoid_fast(accs[2][r] + bf + 1.0f);  // forget bias
                float og = sigmoid_fast(accs[3][r] + bo);
                float cn = c_state[g][r] * fg + ig * jg;
                c_state[g][r] = cn;
                float h = tanh_fast(cn) * og;
                hsum[g][r] += h;
                hwl[r * UPAD + g * 16] = (bf16_t)h;
            }
        }
    }
#undef LDB_N
#undef TILE_G
#undef TILE_GATE
#undef TILE_KC

    // ---------------- mean-pool + ReLU -> LDS (overlay dead W region) ----------------
    __syncthreads();  // all waves done reading sWt
    float* mb = (float*)&sWt[0][0][0] + wv * (16 * HID);
#pragma unroll
    for (int g = 0; g < 7; ++g) {
        const int u = g * 16 + l16;
#pragma unroll
        for (int r = 0; r < 8; ++r) {
            if (u < HID) {
                const int m = half ? (r + 8) : r;
                mb[m * HID + u] = fmaxf(hsum[g][r] * (1.0f / SEQ), 0.0f);
            }
        }
    }
    __syncthreads();

    // ---------------- tiny heads: 16 rows x 12 outputs per wave ----------------
#pragma unroll 1
    for (int q = 0; q < 6; ++q) {
        const int task = lane + 32 * q;          // 0..191
        const int m    = task / 12;
        const int oidx = task % 12;
        const int j    = (oidx < 6) ? oidx : oidx - 6;
        const float* wmat = (oidx < 6) ? W_act : W_obj;
        float s = (oidx < 6) ? b_act[j] : b_obj[j];
        const float* mrow = mb + m * HID;
#pragma unroll 4
        for (int u = 0; u < HID; ++u) s += mrow[u] * wmat[u * 6 + j];
        const size_t base = (oidx < 6) ? 0 : (size_t)B_TOT * 6;
        out[base + (size_t)(b0 + m) * 6 + j] = s;
    }
}

extern "C" void kernel_launch(void* const* d_in, const int* in_sizes, int n_in,
                              void* d_out, int out_size, void* d_ws, size_t ws_size,
                              hipStream_t stream) {
    (void)in_sizes; (void)n_in; (void)out_size; (void)d_ws; (void)ws_size;
    const int*   inputs = (const int*)  d_in[0];
    const float* embed  = (const float*)d_in[1];
    const float* W_lstm = (const float*)d_in[2];
    const float* b_lstm = (const float*)d_in[3];
    const float* W_act  = (const float*)d_in[4];
    const float* b_act  = (const float*)d_in[5];
    const float* W_obj  = (const float*)d_in[6];
    const float* b_obj  = (const float*)d_in[7];
    float* out = (float*)d_out;

    dim3 grid(B_TOT / ROWS_PER_BLOCK);   // 128 blocks
    dim3 block(256);                     // 8 waves of 32
    lstm_dqn_kernel<<<grid, block, 0, stream>>>(inputs, embed, W_lstm, b_lstm,
                                                W_act, b_act, W_obj, b_obj, out);
}
